// Head_62354335203347
// MI455X (gfx1250) — matrix operand order repaired and checked by reading, not compiled, not run on hardware
//
#include <hip/hip_runtime.h>
#include <hip/hip_bf16.h>

// ---------------- WMMA vector types (gfx1250, wave32) ----------------
typedef __attribute__((ext_vector_type(16))) __bf16 v16bf;
typedef __attribute__((ext_vector_type(8)))  __bf16 v8bf;
typedef __attribute__((ext_vector_type(8)))  float  v8f;

#define B_   4
#define T_   2048
#define E_   1024
#define H_   128
#define KT_  (E_/32)      // 32 K-tiles in projection GEMM
#define NT_  (H_/16)      // 8  N-tiles total
#define NTW_ 4            // N-tiles per wave in projection (halved for reg pressure)
#define MT_  ((B_*T_)/16) // 512 M-tiles

// ------------------------------------------------------------------
// Kernel 0: repack Wq/Wk/Wv (fp32 row-major [E][H]) into bf16 WMMA
// B-operand tile layout: tile (w,kt,nt), 32 lanes x 16 contiguous bf16.
// B order used here (contiguous k, not the ISA table order): lane n = lane&15, element i <-> K = (lane>>4)*16 + i.
// ------------------------------------------------------------------
__global__ void pack_w_kernel(const float* __restrict__ Wq,
                              const float* __restrict__ Wk,
                              const float* __restrict__ Wv,
                              __bf16* __restrict__ pW) {
    int gid  = blockIdx.x * blockDim.x + threadIdx.x;
    int lane = gid & 31;
    int tile = gid >> 5;                 // w*KT_*NT_ + kt*NT_ + nt
    if (tile >= 3 * KT_ * NT_) return;
    int w  = tile / (KT_ * NT_);
    int kt = (tile / NT_) % KT_;
    int nt = tile % NT_;
    const float* W = (w == 0) ? Wq : ((w == 1) ? Wk : Wv);
    int khalf = lane >> 4;
    int n = nt * 16 + (lane & 15);
    __bf16* dst = pW + (size_t)tile * 512 + (size_t)lane * 16;
#pragma unroll
    for (int i = 0; i < 16; ++i) {
        int K = kt * 32 + khalf * 16 + i;
        dst[i] = (__bf16)W[(size_t)K * H_ + n];
    }
}

// ------------------------------------------------------------------
// Kernel 1: QKV projection. One wave per (matrix w, M-tile mt, N-half nh).
// Wave computes a 16x64 slab: acc = 4x v8f (32 VGPRs), 4 staged B tiles
// (32 VGPRs) -> ~90 VGPRs total, no spills, 8 b128 loads in flight/step.
// Q,K -> row-major bf16 [B*T][H]; V -> transposed bf16 Vt[B][H][T].
// ------------------------------------------------------------------
__global__ void qkv_gemm_kernel(const float* __restrict__ x,
                                const __bf16* __restrict__ pW,
                                __bf16* __restrict__ Qb,
                                __bf16* __restrict__ Kb,
                                __bf16* __restrict__ Vt) {
    int wid = blockIdx.x * (blockDim.x >> 5) + (threadIdx.x >> 5);
    if (wid >= 3 * MT_ * 2) return;
    int lane  = threadIdx.x & 31;
    int w     = wid / (MT_ * 2);
    int rem   = wid % (MT_ * 2);
    int mt    = rem >> 1;
    int nh    = rem & 1;        // which half of H: N-tile base = nh*4
    int m     = lane & 15;      // A-layout row within tile
    int khalf = lane >> 4;

    const float* xr = x + (size_t)(mt * 16 + m) * E_;
    const __bf16* pw = pW + (size_t)w * (KT_ * NT_ * 512);

    v8f acc[NTW_];
#pragma unroll
    for (int j = 0; j < NTW_; ++j)
#pragma unroll
        for (int r = 0; r < 8; ++r) acc[j][r] = 0.f;

    for (int kt = 0; kt < KT_; ++kt) {
        const float* base = xr + kt * 32;
        if (kt + 1 < KT_) __builtin_prefetch(base + 32 + khalf * 8, 0, 0);

        // A (16x32 bf16): element i <-> K = khalf*16 + i
        v16bf a;
#pragma unroll
        for (int g = 0; g < 2; ++g) {
            const float4* p = (const float4*)(base + khalf * 16 + g * 8);
            float4 f0 = p[0], f1 = p[1];
            int o = g * 8;
            a[o+0]=(__bf16)f0.x; a[o+1]=(__bf16)f0.y; a[o+2]=(__bf16)f0.z; a[o+3]=(__bf16)f0.w;
            a[o+4]=(__bf16)f1.x; a[o+5]=(__bf16)f1.y; a[o+6]=(__bf16)f1.z; a[o+7]=(__bf16)f1.w;
        }
        // stage this wave's 4 B-operand tiles, then run the WMMA chain
        v16bf bt[NTW_];
#pragma unroll
        for (int j = 0; j < NTW_; ++j)
            bt[j] = *(const v16bf*)(pw + (size_t)(kt * NT_ + nh * NTW_ + j) * 512
                                       + (size_t)lane * 16);
#pragma unroll
        for (int j = 0; j < NTW_; ++j)
            acc[j] = __builtin_amdgcn_wmma_f32_16x16x32_bf16(
                false, a, false, bt[j], (short)0, acc[j], false, false);
    }

    // C layout: column n = lane&15, row M = r + 8*khalf
    if (w < 2) {
        __bf16* out = (w == 0) ? Qb : Kb;
#pragma unroll
        for (int j = 0; j < NTW_; ++j) {
            int nt = nh * NTW_ + j;
#pragma unroll
            for (int r = 0; r < 8; ++r) {
                int M = r + 8 * khalf;
                out[(size_t)(mt * 16 + M) * H_ + nt * 16 + m] = (__bf16)acc[j][r];
            }
        }
    } else {
#pragma unroll
        for (int j = 0; j < NTW_; ++j) {
            int nt = nh * NTW_ + j;
#pragma unroll
            for (int r = 0; r < 8; ++r) {
                int M    = r + 8 * khalf;
                int rowg = mt * 16 + M;
                int bb   = rowg >> 11;         // /T_
                int t    = rowg & (T_ - 1);
                Vt[((size_t)bb * H_ + nt * 16 + m) * T_ + t] = (__bf16)acc[j][r];
            }
        }
    }
}

// ------------------------------------------------------------------
// Kernel 2: fused causal attention (flash style).
// 2 waves per 16-query tile; wave w handles key blocks kb = 32w + 64k.
// Partial (m, l, O) merged via LDS with the standard flash combine.
// Operands are streamed (not staged) to stay below the spill threshold:
// live set ~= qa(32) + acc(64) + transient(16) VGPRs.
// ------------------------------------------------------------------
__global__ void attn_kernel(const __bf16* __restrict__ Qb,
                            const __bf16* __restrict__ Kb,
                            const __bf16* __restrict__ Vt,
                            float* __restrict__ out) {
    __shared__ __attribute__((aligned(32))) __bf16 p_lds[2][16 * 32]; // per-wave P staging
    __shared__ float red_ml[32 * 16];     // wave1 m[8], l[8] per lane
    __shared__ float red_acc[32 * 64];    // wave1 acc[8][8] per lane

    int wid   = blockIdx.x;            // 0 .. B_*T_/16 - 1
    int lane  = threadIdx.x & 31;
    int wv    = threadIdx.x >> 5;      // 0 or 1: key-range split
    int b     = wid >> 7;              // T_/16 = 128 tiles per batch
    int qt    = wid & 127;
    int q0    = qt * 16;
    int m     = lane & 15;
    int khalf = lane >> 4;
    const float scale = 0.08838834764831845f;  // 1/sqrt(128)

    // Q tile in A-layout registers: 4 chunks of 16x32
    v16bf qa[4];
    const __bf16* qrow = Qb + ((size_t)b * T_ + q0 + m) * H_;
#pragma unroll
    for (int c = 0; c < 4; ++c)
#pragma unroll
        for (int g = 0; g < 2; ++g) {
            v8bf t = *(const v8bf*)(qrow + c * 32 + khalf * 16 + g * 8);
#pragma unroll
            for (int j = 0; j < 8; ++j) qa[c][g * 8 + j] = t[j];
        }

    v8f acc[NT_];
    float mprev[8], lsum[8];
#pragma unroll
    for (int r = 0; r < 8; ++r) { mprev[r] = -__builtin_inff(); lsum[r] = 0.f; }
#pragma unroll
    for (int nt = 0; nt < NT_; ++nt)
#pragma unroll
        for (int r = 0; r < 8; ++r) acc[nt][r] = 0.f;

    const int kend = q0 + 16;
    for (int kb = wv * 32; kb < kend; kb += 64) {
        // prefetch this wave's next key block (K rows + a slice of V rows)
        int nkb = kb + 64;
        if (nkb < kend) {
            __builtin_prefetch(Kb + ((size_t)b * T_ + nkb + m)      * H_ + khalf * 64, 0, 0);
            __builtin_prefetch(Kb + ((size_t)b * T_ + nkb + 16 + m) * H_ + khalf * 64, 0, 0);
            __builtin_prefetch(Vt + ((size_t)b * H_ + lane * 4)     * T_ + nkb, 0, 0);
        }

        // ---- S = Q * K^T for 32 keys (streamed B loads) ----
        v8f s0, s1;
#pragma unroll
        for (int r = 0; r < 8; ++r) { s0[r] = 0.f; s1[r] = 0.f; }
#pragma unroll
        for (int c = 0; c < 4; ++c) {
            v16bf k0 = *(const v16bf*)(Kb + ((size_t)b * T_ + kb + m)      * H_ + c * 32 + khalf * 16);
            v16bf k1 = *(const v16bf*)(Kb + ((size_t)b * T_ + kb + 16 + m) * H_ + c * 32 + khalf * 16);
            s0 = __builtin_amdgcn_wmma_f32_16x16x32_bf16(false, qa[c], false, k0, (short)0, s0, false, false);
            s1 = __builtin_amdgcn_wmma_f32_16x16x32_bf16(false, qa[c], false, k1, (short)0, s1, false, false);
        }

        // ---- online softmax (rows in element r + 8*khalf; cols in lanes) ----
#pragma unroll
        for (int r = 0; r < 8; ++r) {
            int q  = q0 + r + 8 * khalf;
            float v0 = s0[r] * scale;
            float v1 = s1[r] * scale;
            if (kb + m      > q) v0 = -__builtin_inff();
            if (kb + 16 + m > q) v1 = -__builtin_inff();

            float mx = fmaxf(v0, v1);
            mx = fmaxf(mx, __shfl_xor(mx, 1, 32));
            mx = fmaxf(mx, __shfl_xor(mx, 2, 32));
            mx = fmaxf(mx, __shfl_xor(mx, 4, 32));
            mx = fmaxf(mx, __shfl_xor(mx, 8, 32));

            float mnew  = fmaxf(mprev[r], mx);
            float alpha = __expf(mprev[r] - mnew);
            v0 = __expf(v0 - mnew);
            v1 = __expf(v1 - mnew);

            float rs = v0 + v1;
            rs += __shfl_xor(rs, 1, 32);
            rs += __shfl_xor(rs, 2, 32);
            rs += __shfl_xor(rs, 4, 32);
            rs += __shfl_xor(rs, 8, 32);

            lsum[r]  = lsum[r] * alpha + rs;
            mprev[r] = mnew;
#pragma unroll
            for (int nt = 0; nt < NT_; ++nt) acc[nt][r] *= alpha;

            // C->A transpose staging: P[row][key] bf16 (same-wave LDS only)
            p_lds[wv][(r + 8 * khalf) * 32 + m]      = (__bf16)v0;
            p_lds[wv][(r + 8 * khalf) * 32 + 16 + m] = (__bf16)v1;
        }
        // same-wave DS ordering: drain DS writes, block compiler reordering
        asm volatile("s_wait_dscnt 0x0" ::: "memory");

        // ---- read P in A layout: row m, element i <-> key khalf*16 + i ----
        v16bf pa;
#pragma unroll
        for (int g = 0; g < 2; ++g) {
            v8bf t = *(const v8bf*)(&p_lds[wv][0] + m * 32 + khalf * 16 + g * 8);
#pragma unroll
            for (int j = 0; j < 8; ++j) pa[g * 8 + j] = t[j];
        }
        asm volatile("" ::: "memory");  // keep reads before next iteration's writes

        // ---- O += P(16x32) * V(32x16) (streamed B loads) ----
#pragma unroll
        for (int nt = 0; nt < NT_; ++nt) {
            v16bf bv = *(const v16bf*)(Vt + ((size_t)b * H_ + nt * 16 + m) * T_ + kb + khalf * 16);
            acc[nt] = __builtin_amdgcn_wmma_f32_16x16x32_bf16(
                false, pa, false, bv, (short)0, acc[nt], false, false);
        }
    }

    // ---- merge the two waves' partial states (flash combine), then write ----
    if (wv == 1) {
#pragma unroll
        for (int r = 0; r < 8; ++r) {
            red_ml[lane * 16 + r]     = mprev[r];
            red_ml[lane * 16 + 8 + r] = lsum[r];
        }
#pragma unroll
        for (int nt = 0; nt < NT_; ++nt)
#pragma unroll
            for (int r = 0; r < 8; ++r)
                red_acc[lane * 64 + nt * 8 + r] = acc[nt][r];
    }
    __syncthreads();
    if (wv == 0) {
#pragma unroll
        for (int r = 0; r < 8; ++r) {
            float m1 = red_ml[lane * 16 + r];
            float l1 = red_ml[lane * 16 + 8 + r];
            float M  = fmaxf(mprev[r], m1);           // mprev[r] always finite (key 0)
            float a0 = __expf(mprev[r] - M);
            float a1 = __expf(m1 - M);                // exp(-inf) = 0 when wave1 idle
            float L  = lsum[r] * a0 + l1 * a1;
            float inv = 1.0f / L;
            int   row = q0 + r + 8 * khalf;
#pragma unroll
            for (int nt = 0; nt < NT_; ++nt) {
                float o = (acc[nt][r] * a0 + red_acc[lane * 64 + nt * 8 + r] * a1) * inv;
                out[((size_t)b * T_ + row) * H_ + nt * 16 + m] = o;
            }
        }
    }
}

// ------------------------------------------------------------------
extern "C" void kernel_launch(void* const* d_in, const int* in_sizes, int n_in,
                              void* d_out, int out_size, void* d_ws, size_t ws_size,
                              hipStream_t stream) {
    const float* x  = (const float*)d_in[0];
    const float* Wq = (const float*)d_in[1];
    const float* Wk = (const float*)d_in[2];
    const float* Wv = (const float*)d_in[3];
    float* out = (float*)d_out;

    // workspace layout (all 256B-aligned offsets)
    char* ws = (char*)d_ws;
    const size_t PW_BYTES  = (size_t)3 * KT_ * NT_ * 512 * 2;  // 768 KB packed weights
    const size_t QKV_BYTES = (size_t)B_ * T_ * H_ * 2;         // 2 MB each
    __bf16* pW = (__bf16*)ws;
    __bf16* Qb = (__bf16*)(ws + PW_BYTES);
    __bf16* Kb = (__bf16*)(ws + PW_BYTES + QKV_BYTES);
    __bf16* Vt = (__bf16*)(ws + PW_BYTES + 2 * QKV_BYTES);

    // Kernel 0: pack weights (3*32*8 tiles * 32 lanes = 24576 threads)
    {
        int threads = 3 * KT_ * NT_ * 32;
        pack_w_kernel<<<(threads + 255) / 256, 256, 0, stream>>>(Wq, Wk, Wv, pW);
    }
    // Kernel 1: QKV projection (3*512*2 waves, 4 waves per 128-thread block)
    {
        int waves = 3 * MT_ * 2;
        qkv_gemm_kernel<<<waves / 4, 128, 0, stream>>>(x, pW, Qb, Kb, Vt);
    }
    // Kernel 2: fused causal attention (512 blocks x 2 waves, split key range)
    {
        attn_kernel<<<B_ * (T_ / 16), 64, 0, stream>>>(Qb, Kb, Vt, out);
    }
}
